// SlidingWindowAttention_78297253806473
// MI455X (gfx1250) — compile-verified, matrix operand order checked by reading, not run on hardware
//
#include <hip/hip_runtime.h>

// ---------------------------------------------------------------------------
// Sliding-window attention for MI455X (gfx1250), all matmuls via
// v_wmma_f32_16x16x32_bf16. wave32.
// Shapes: B=2, S=2048, HID=1024, H=16 heads, D=64, window=512 (+-256).
// ---------------------------------------------------------------------------

#define B_ 2
#define S_ 2048
#define HID_ 1024
#define H_ 16
#define D_ 64
#define HALFW 256
#define SCALING 0.125f   // 64^-0.5

typedef unsigned short u16t;
typedef __attribute__((ext_vector_type(8)))  __bf16 v8bf;
typedef __attribute__((ext_vector_type(16))) __bf16 v16bf;
typedef __attribute__((ext_vector_type(8)))  float  v8f;

__device__ __forceinline__ v8f wmma_bf16(v16bf a, v16bf b, v8f c) {
    return __builtin_amdgcn_wmma_f32_16x16x32_bf16(
        /*neg_a=*/false, a, /*neg_b=*/false, b,
        /*c_mod=*/(short)0, c, /*reuse_a=*/false, /*reuse_b=*/false);
}

__device__ __forceinline__ v8f zero8() {
    v8f z;
#pragma unroll
    for (int i = 0; i < 8; ++i) z[i] = 0.0f;
    return z;
}

__device__ __forceinline__ unsigned short f32_to_bf16(float f) {
    unsigned u = __builtin_bit_cast(unsigned, f);
    u += 0x7FFFu + ((u >> 16) & 1u);   // round to nearest even
    return (unsigned short)(u >> 16);
}

// Load one 16x32 bf16 A-fragment (or the symmetric B-fragment from a
// row-major [N,K] matrix). Per ISA layout: lane L holds row (L&15); k-chunks
// at kbase = 8*(L>>4) and kbase+16, each 8 contiguous bf16 (16 bytes).
__device__ __forceinline__ v16bf load_frag(const __bf16* base, int stride, int lane) {
    const int row = lane & 15;
    const int kb  = (lane >> 4) << 3;
    const __bf16* p = base + (size_t)row * stride + kb;
    v8bf lo = *(const v8bf*)(p);
    v8bf hi = *(const v8bf*)(p + 16);
    v16bf r;
#pragma unroll
    for (int i = 0; i < 8; ++i) { r[i] = lo[i]; r[i + 8] = hi[i]; }
    return r;
}

// ---------------------------------------------------------------------------
// fp32 -> bf16 conversion (4 elements / thread)
// ---------------------------------------------------------------------------
__global__ void cvt_bf16_kernel(const float* __restrict__ src,
                                u16t* __restrict__ dst, int n4) {
    int i = blockIdx.x * blockDim.x + threadIdx.x;
    if (i >= n4) return;
    float4 f = ((const float4*)src)[i];
    unsigned short a = f32_to_bf16(f.x), b = f32_to_bf16(f.y);
    unsigned short c = f32_to_bf16(f.z), d = f32_to_bf16(f.w);
    uint2 v;
    v.x = (unsigned)a | ((unsigned)b << 16);
    v.y = (unsigned)c | ((unsigned)d << 16);
    ((uint2*)dst)[i] = v;
}

// ---------------------------------------------------------------------------
// Zero attn buffer (float4 / thread)
// ---------------------------------------------------------------------------
__global__ void zero_f4_kernel(float4* __restrict__ p, int n4) {
    int i = blockIdx.x * blockDim.x + threadIdx.x;
    if (i < n4) p[i] = make_float4(0.f, 0.f, 0.f, 0.f);
}

// ---------------------------------------------------------------------------
// 32x64 output tile per wave: acc = X[m0:m0+32, :] * W[n0:n0+64, :]^T
// X row-major [M,K] bf16, W row-major [N,K] bf16, K = 1024.
// ---------------------------------------------------------------------------
__device__ __forceinline__ void gemm_tile_32x64(const __bf16* X, const __bf16* W,
                                                int m0, int n0, int lane,
                                                v8f acc[2][4]) {
#pragma unroll
    for (int ms = 0; ms < 2; ++ms)
#pragma unroll
        for (int nt = 0; nt < 4; ++nt) acc[ms][nt] = zero8();

    for (int kk = 0; kk < HID_; kk += 32) {
        v16bf a0 = load_frag(X + (size_t)(m0     ) * HID_ + kk, HID_, lane);
        v16bf a1 = load_frag(X + (size_t)(m0 + 16) * HID_ + kk, HID_, lane);
        v16bf b0 = load_frag(W + (size_t)(n0     ) * HID_ + kk, HID_, lane);
        v16bf b1 = load_frag(W + (size_t)(n0 + 16) * HID_ + kk, HID_, lane);
        v16bf b2 = load_frag(W + (size_t)(n0 + 32) * HID_ + kk, HID_, lane);
        v16bf b3 = load_frag(W + (size_t)(n0 + 48) * HID_ + kk, HID_, lane);
        acc[0][0] = wmma_bf16(a0, b0, acc[0][0]);
        acc[0][1] = wmma_bf16(a0, b1, acc[0][1]);
        acc[0][2] = wmma_bf16(a0, b2, acc[0][2]);
        acc[0][3] = wmma_bf16(a0, b3, acc[0][3]);
        acc[1][0] = wmma_bf16(a1, b0, acc[1][0]);
        acc[1][1] = wmma_bf16(a1, b1, acc[1][1]);
        acc[1][2] = wmma_bf16(a1, b2, acc[1][2]);
        acc[1][3] = wmma_bf16(a1, b3, acc[1][3]);
    }
}

// ---------------------------------------------------------------------------
// Q/K/V projection. grid.z = 0/1/2 selects W and destination layout.
//   Q,K -> bf16 [B,H,S,D]   V -> bf16 [B,H,D,S] (transposed for P@V frags)
// ---------------------------------------------------------------------------
__global__ __launch_bounds__(256)
void proj_qkv_kernel(const u16t* __restrict__ xraw,
                     const u16t* __restrict__ wqr, const u16t* __restrict__ wkr,
                     const u16t* __restrict__ wvr,
                     u16t* __restrict__ qws, u16t* __restrict__ kws,
                     u16t* __restrict__ vtws) {
    const int mode = blockIdx.z;
    const __bf16* X = (const __bf16*)xraw;
    const __bf16* W = (const __bf16*)(mode == 0 ? wqr : (mode == 1 ? wkr : wvr));
    u16t* dst       = (mode == 0) ? qws : (mode == 1 ? kws : vtws);

    const int lane = threadIdx.x & 31;
    const int wid  = blockIdx.x * 8 + (threadIdx.x >> 5);   // 0..2047
    const int m0 = (wid >> 4) * 32;
    const int n0 = (wid & 15) * 64;

    v8f acc[2][4];
    gemm_tile_32x64(X, W, m0, n0, lane, acc);

    const int n_  = lane & 15;
    const int hig = lane >> 4;
#pragma unroll
    for (int ms = 0; ms < 2; ++ms)
#pragma unroll
        for (int nt = 0; nt < 4; ++nt)
#pragma unroll
            for (int r = 0; r < 8; ++r) {
                int m  = m0 + ms * 16 + r + 8 * hig;   // row in [0, B*S)
                int nn = n0 + nt * 16 + n_;            // col in [0, HID)
                int b = m >> 11, s = m & (S_ - 1);
                int h = nn >> 6, d = nn & (D_ - 1);
                unsigned short hv = f32_to_bf16(acc[ms][nt][r]);
                size_t idx = (mode == 2)
                    ? ((size_t)((b * H_ + h) * D_ + d) * S_ + s)     // Vt
                    : ((size_t)((b * H_ + h) * S_ + s) * D_ + d);    // Q/K
                dst[idx] = hv;
            }
}

// ---------------------------------------------------------------------------
// Attention: one wave handles a 16-row tile of one (b,h).
// Pass 1: online (rowmax, rowsum) over the window column tiles.
// Pass 2: normalized probs -> attn output + P@V via LDS-transposed A-frags.
// ---------------------------------------------------------------------------
__global__ __launch_bounds__(256)
void attn_kernel(const u16t* __restrict__ qraw, const u16t* __restrict__ kraw,
                 const u16t* __restrict__ vtraw,
                 float* __restrict__ attn, u16t* __restrict__ ctxraw) {
    __shared__ __bf16 ldsAll[8 * 16 * 32];   // 8 waves x (16x32 bf16) = 8KB
    const int lane = threadIdx.x & 31;
    const int warp = threadIdx.x >> 5;
    __bf16* lds = ldsAll + warp * 512;

    const int wid = blockIdx.x * 8 + warp;   // 0..4095
    const int b  = wid >> 11;
    const int h  = (wid >> 7) & (H_ - 1);
    const int i0 = (wid & 127) * 16;

    const __bf16* qb  = (const __bf16*)qraw  + (size_t)(b * H_ + h) * S_ * D_;
    const __bf16* kb  = (const __bf16*)kraw  + (size_t)(b * H_ + h) * S_ * D_;
    const __bf16* vtb = (const __bf16*)vtraw + (size_t)(b * H_ + h) * D_ * S_;
    float* ao = attn + (size_t)(b * H_ + h) * S_ * S_ + (size_t)i0 * S_;

    const v16bf aq0 = load_frag(qb + (size_t)i0 * D_,      D_, lane);
    const v16bf aq1 = load_frag(qb + (size_t)i0 * D_ + 32, D_, lane);

    int jlo = i0 - HALFW;      if (jlo < 0) jlo = 0;
    int jhi = i0 + 15 + HALFW; if (jhi > S_ - 1) jhi = S_ - 1;
    const int t0 = jlo >> 4, t1 = jhi >> 4;

    const int n_  = lane & 15;
    const int hig = lane >> 4;

    float mx[8], sm[8];
#pragma unroll
    for (int r = 0; r < 8; ++r) { mx[r] = -1e30f; sm[r] = 0.0f; }

    // ---- pass 1: online max/sum ----
    for (int t = t0; t <= t1; ++t) {
        const int j0 = t << 4;
        v16bf bk0 = load_frag(kb + (size_t)j0 * D_,      D_, lane);
        v16bf bk1 = load_frag(kb + (size_t)j0 * D_ + 32, D_, lane);
        v8f s = zero8();
        s = wmma_bf16(aq0, bk0, s);
        s = wmma_bf16(aq1, bk1, s);
        const int j = j0 + n_;
#pragma unroll
        for (int r = 0; r < 8; ++r) {
            int i = i0 + r + 8 * hig;
            float v = s[r] * SCALING;
            int di = i - j; if (di < 0) di = -di;
            v = (di <= HALFW) ? v : -1e30f;
            float nm = fmaxf(mx[r], v);
            sm[r] = sm[r] * __expf(mx[r] - nm) + __expf(v - nm);
            mx[r] = nm;
        }
    }
    // merge (max,sum) across the 16-lane group (row spread over lanes)
#pragma unroll
    for (int xm = 1; xm < 16; xm <<= 1) {
#pragma unroll
        for (int r = 0; r < 8; ++r) {
            float om = __shfl_xor(mx[r], xm, 32);
            float os = __shfl_xor(sm[r], xm, 32);
            float nm = fmaxf(mx[r], om);
            sm[r] = sm[r] * __expf(mx[r] - nm) + os * __expf(om - nm);
            mx[r] = nm;
        }
    }
    float inv[8];
#pragma unroll
    for (int r = 0; r < 8; ++r) inv[r] = 1.0f / sm[r];

    // ---- pass 2: normalized probs -> attn store + P@V ----
    v8f o[4];
#pragma unroll
    for (int nt = 0; nt < 4; ++nt) o[nt] = zero8();

    const int npair = ((t1 - t0 + 1) + 1) >> 1;
    for (int pt = 0; pt < npair; ++pt) {
#pragma unroll
        for (int sub = 0; sub < 2; ++sub) {
            const int t = t0 + pt * 2 + sub;
            if (t <= t1) {
                const int j0 = t << 4;
                v16bf bk0 = load_frag(kb + (size_t)j0 * D_,      D_, lane);
                v16bf bk1 = load_frag(kb + (size_t)j0 * D_ + 32, D_, lane);
                v8f s = zero8();
                s = wmma_bf16(aq0, bk0, s);
                s = wmma_bf16(aq1, bk1, s);
                const int j = j0 + n_;
#pragma unroll
                for (int r = 0; r < 8; ++r) {
                    int i = i0 + r + 8 * hig;
                    float v = s[r] * SCALING;
                    int di = i - j; if (di < 0) di = -di;
                    v = (di <= HALFW) ? v : -1e30f;
                    float p = __expf(v - mx[r]) * inv[r];
                    ao[(size_t)(r + 8 * hig) * S_ + j] = p;
                    lds[(r + 8 * hig) * 32 + sub * 16 + n_] =
                        __builtin_bit_cast(__bf16, f32_to_bf16(p));
                }
            } else {  // zero-pad second half of the LDS pair
#pragma unroll
                for (int r = 0; r < 8; ++r)
                    lds[(r + 8 * hig) * 32 + 16 + n_] =
                        __builtin_bit_cast(__bf16, (unsigned short)0);
            }
        }
        asm volatile("s_wait_dscnt 0" ::: "memory");
        v16bf ap = load_frag(lds, 32, lane);   // P in A-layout (16x32)
        const int j0p = (t0 + pt * 2) << 4;
#pragma unroll
        for (int nt = 0; nt < 4; ++nt) {
            v16bf bv = load_frag(vtb + (size_t)(nt * 16) * S_ + j0p, S_, lane);
            o[nt] = wmma_bf16(ap, bv, o[nt]);
        }
    }

    // store context [B,S,HID] bf16 (already normalized)
#pragma unroll
    for (int nt = 0; nt < 4; ++nt)
#pragma unroll
        for (int r = 0; r < 8; ++r) {
            int m = i0 + r + 8 * hig;
            int d = nt * 16 + n_;
            ctxraw[(size_t)(b * S_ + m) * HID_ + h * D_ + d] =
                f32_to_bf16(o[nt][r]);
        }
}

// ---------------------------------------------------------------------------
// Output projection: out = ctx @ Wo^T, fp32 stores into d_out.
// ---------------------------------------------------------------------------
__global__ __launch_bounds__(256)
void outproj_kernel(const u16t* __restrict__ ctxraw, const u16t* __restrict__ wor,
                    float* __restrict__ out) {
    const __bf16* X = (const __bf16*)ctxraw;
    const __bf16* W = (const __bf16*)wor;
    const int lane = threadIdx.x & 31;
    const int wid  = blockIdx.x * 8 + (threadIdx.x >> 5);
    const int m0 = (wid >> 4) * 32;
    const int n0 = (wid & 15) * 64;

    v8f acc[2][4];
    gemm_tile_32x64(X, W, m0, n0, lane, acc);

    const int n_  = lane & 15;
    const int hig = lane >> 4;
#pragma unroll
    for (int ms = 0; ms < 2; ++ms)
#pragma unroll
        for (int nt = 0; nt < 4; ++nt)
#pragma unroll
            for (int r = 0; r < 8; ++r) {
                int m  = m0 + ms * 16 + r + 8 * hig;
                int nn = n0 + nt * 16 + n_;
                out[(size_t)m * HID_ + nn] = acc[ms][nt][r];
            }
}

// ---------------------------------------------------------------------------
extern "C" void kernel_launch(void* const* d_in, const int* in_sizes, int n_in,
                              void* d_out, int out_size, void* d_ws, size_t ws_size,
                              hipStream_t stream) {
    const float* hidden = (const float*)d_in[0];   // [2,2048,1024]
    const float* Wq = (const float*)d_in[1];       // [1024,1024]
    const float* Wk = (const float*)d_in[2];
    const float* Wv = (const float*)d_in[3];
    const float* Wo = (const float*)d_in[4];

    float* out  = (float*)d_out;                              // [2,2048,1024]
    float* attn = out + (size_t)B_ * S_ * HID_;               // [2,16,2048,2048]

    char* ws = (char*)d_ws;
    const size_t MB = 1ull << 20;
    u16t* xbf  = (u16t*)(ws);                 // 8 MB  [B*S, HID] bf16
    u16t* wqb  = (u16t*)(ws +  8 * MB);       // 2 MB
    u16t* wkb  = (u16t*)(ws + 10 * MB);
    u16t* wvb  = (u16t*)(ws + 12 * MB);
    u16t* wob  = (u16t*)(ws + 14 * MB);
    u16t* qws  = (u16t*)(ws + 16 * MB);       // 8 MB  [B,H,S,D]
    u16t* kws  = (u16t*)(ws + 24 * MB);       // 8 MB  [B,H,S,D]
    u16t* vtws = (u16t*)(ws + 32 * MB);       // 8 MB  [B,H,D,S]
    u16t* ctx  = (u16t*)(ws + 40 * MB);       // 8 MB  [B,S,HID]

    // 1) fp32 -> bf16
    cvt_bf16_kernel<<<4096, 256, 0, stream>>>(hidden, xbf, (B_ * S_ * HID_) / 4);
    cvt_bf16_kernel<<<1024, 256, 0, stream>>>(Wq, wqb, (HID_ * HID_) / 4);
    cvt_bf16_kernel<<<1024, 256, 0, stream>>>(Wk, wkb, (HID_ * HID_) / 4);
    cvt_bf16_kernel<<<1024, 256, 0, stream>>>(Wv, wvb, (HID_ * HID_) / 4);
    cvt_bf16_kernel<<<1024, 256, 0, stream>>>(Wo, wob, (HID_ * HID_) / 4);

    // 2) Q/K/V projections (2048 waves per projection, 8 waves/block)
    proj_qkv_kernel<<<dim3(256, 1, 3), 256, 0, stream>>>(
        xbf, wqb, wkb, wvb, qws, kws, vtws);

    // 3) zero attn output (masked region must be exactly 0)
    zero_f4_kernel<<<131072, 256, 0, stream>>>(
        (float4*)attn, (int)((size_t)B_ * H_ * S_ * S_ / 4));

    // 4) attention (4096 waves, one 16-row tile each)
    attn_kernel<<<512, 256, 0, stream>>>(qws, kws, vtws, attn, ctx);

    // 5) output projection
    outproj_kernel<<<256, 256, 0, stream>>>(ctx, wob, out);
}
